// StereoINR_45500883534288
// MI455X (gfx1250) — matrix operand order repaired and checked by reading, not compiled, not run on hardware
//
#include <hip/hip_runtime.h>
#include <hip/hip_bf16.h>

// ---------------------------------------------------------------------------
// StereoINR fused implementation for gfx1250 (MI455X)
//   N=2, C=64, H=W=96, Q=147456, HD=128
// WMMA f16 MLPs with TDM (tensor_load_to_lds) weight-panel streaming.
// ---------------------------------------------------------------------------

typedef _Float16 half_t;
typedef __attribute__((ext_vector_type(16))) _Float16 v16h;
typedef __attribute__((ext_vector_type(2)))  _Float16 hvec2;
typedef __attribute__((ext_vector_type(8)))  float    v8f;
typedef __attribute__((ext_vector_type(4)))  unsigned int u32x4;
typedef __attribute__((ext_vector_type(8)))  unsigned int u32x8;

#define NB   2
#define CIN  64
#define HH   96
#define WW   96
#define HW   9216
#define QQ   147456
#define HD_  128

__device__ __forceinline__ float clampf(float v, float lo, float hi) {
    return fminf(fmaxf(v, lo), hi);
}

__device__ __forceinline__ unsigned int lds_off(const void* p) {
    // Low 32 bits of a generic pointer to LDS == LDS byte address (aperture in high bits).
    return (unsigned int)(size_t)p;
}

// ---------------------------------------------------------------------------
// TDM: DMA a K x 16 (f16) weight panel from global into LDS (packed).
// Descriptor per CDNA5 ISA 08_async_tensor §8.3-8.6:
//   data_size = 8B units; tile_dim0 = 4 units (32B = 16 halves per row);
//   tile_dim1 = Krows; tensor_dim0(_stride) = strideU8 (row pitch in 8B units).
// Issued by one wave; tracked with TENSORcnt.
// ---------------------------------------------------------------------------
__device__ __forceinline__ void tdm_panel_load(const half_t* gsrc, unsigned int lds_byte,
                                               int Krows, unsigned int strideU8) {
    unsigned long long ga = (unsigned long long)(size_t)gsrc;
    u32x4 g0;
    g0[0] = 1u;                                        // count=1, user mode
    g0[1] = lds_byte;                                  // lds_addr
    g0[2] = (unsigned int)ga;                          // global_addr[31:0]
    g0[3] = (unsigned int)(ga >> 32) | (2u << 30);     // global_addr[56:32] | type=2
    const unsigned int T0 = strideU8;                  // tensor_dim0 (units)
    const unsigned int T1 = (unsigned int)Krows;       // tensor_dim1
    u32x8 g1;
    g1[0] = (3u << 16);                                // data_size = 8B
    g1[1] = (T0 & 0xffffu) << 16;                      // tensor_dim0[15:0] @ bits 63:48
    g1[2] = ((T0 >> 16) & 0xffffu) | ((T1 & 0xffffu) << 16);   // dim0 hi | dim1 lo
    g1[3] = ((T1 >> 16) & 0xffffu) | (4u << 16);       // dim1 hi | tile_dim0 = 4
    g1[4] = (T1 & 0xffffu);                            // tile_dim1 = Krows; tile_dim2 = 0
    g1[5] = T0;                                        // tensor_dim0_stride[31:0]
    g1[6] = 0u;                                        // stride0 hi | stride1 lo
    g1[7] = 0u;
    asm volatile("tensor_load_to_lds %0, %1" :: "s"(g0), "s"(g1) : "memory");
}

// ---------------------------------------------------------------------------
// K0: pack weights to f16 with padding.
// ---------------------------------------------------------------------------
__global__ void prep_weights(const float* im_w1, const float* im_w2, const float* im_w3,
                             const float* dp_w1, const float* dp_w2, const float* dp_w3,
                             half_t* w1h, half_t* w2h, half_t* w3h,
                             half_t* d1h, half_t* d2h, half_t* d3h) {
    int tid = blockIdx.x * blockDim.x + threadIdx.x;
    int stride = gridDim.x * blockDim.x;
    for (int i = tid; i < 128 * 256; i += stride) w1h[i] = (half_t)im_w1[i];
    for (int i = tid; i < 256 * 256; i += stride) w2h[i] = (half_t)im_w2[i];
    for (int i = tid; i < 256 * 16; i += stride) {
        int k = i >> 4, nn = i & 15;
        w3h[i] = (nn < 3) ? (half_t)im_w3[k * 3 + nn] : (half_t)0.0f;
    }
    for (int i = tid; i < 96 * 256; i += stride) {
        int k = i >> 8, nn = i & 255;
        d1h[i] = (k < 67) ? (half_t)dp_w1[k * 256 + nn] : (half_t)0.0f;
    }
    for (int i = tid; i < 256 * 256; i += stride) d2h[i] = (half_t)dp_w2[i];
    for (int i = tid; i < 256 * 16; i += stride) {
        int k = i >> 4, nn = i & 15;
        d3h[i] = (nn < 1) ? (half_t)dp_w3[k] : (half_t)0.0f;
    }
}

// ---------------------------------------------------------------------------
// K1: direct 3x3 SAME conv: feat(N,64,96,96) -> coef/freq (N,128,96,96) fp32
// ---------------------------------------------------------------------------
__global__ void conv3x3(const float* __restrict__ feat,
                        const float* __restrict__ coef_w, const float* __restrict__ coef_b,
                        const float* __restrict__ freq_w, const float* __restrict__ freq_b,
                        float* __restrict__ coef_o, float* __restrict__ freq_o) {
    const int pix = blockIdx.x * blockDim.x + threadIdx.x;   // 0..9215
    const int co  = blockIdx.y;
    const int n   = blockIdx.z;
    const int y = pix / WW, x = pix % WW;
    float accC = coef_b[co];
    float accF = freq_b[co];
    for (int c = 0; c < CIN; ++c) {
        const float* fp = feat + ((size_t)(n * CIN + c)) * HW;
        const float* wc = coef_w + ((size_t)(co * CIN + c)) * 9;
        const float* wf = freq_w + ((size_t)(co * CIN + c)) * 9;
#pragma unroll
        for (int dy = -1; dy <= 1; ++dy) {
            int yy = y + dy;
            if (yy < 0 || yy >= HH) continue;
#pragma unroll
            for (int dx = -1; dx <= 1; ++dx) {
                int xx = x + dx;
                if (xx < 0 || xx >= WW) continue;
                float v = fp[yy * WW + xx];
                int t = (dy + 1) * 3 + (dx + 1);
                accC = fmaf(v, wc[t], accC);
                accF = fmaf(v, wf[t], accF);
            }
        }
    }
    coef_o[((size_t)(n * HD_ + co)) * HW + pix] = accC;
    freq_o[((size_t)(n * HD_ + co)) * HW + pix] = accF;
}

// ---------------------------------------------------------------------------
// K2: disp scale + mean/std(ddof=1) normalization. One block per batch image.
// ---------------------------------------------------------------------------
__global__ void disp_stats(const float* __restrict__ raw_disp, const float* __restrict__ cell,
                           float* __restrict__ disp_o, float* __restrict__ dnorm_o) {
    __shared__ float red[256], red2[256], bc[2];
    const int n = blockIdx.x;
    const int tid = threadIdx.x;
    const float scale = 2.0f / (cell[((size_t)n * QQ) * 2 + 1] + 1e-6f) / (float)WW;
    float s = 0.f, s2 = 0.f;
    for (int i = tid; i < HW; i += 256) {
        float v = raw_disp[(size_t)n * HW + i] * scale;
        s += v; s2 += v * v;
    }
    red[tid] = s; red2[tid] = s2;
    __syncthreads();
    for (int off = 128; off > 0; off >>= 1) {
        if (tid < off) { red[tid] += red[tid + off]; red2[tid] += red2[tid + off]; }
        __syncthreads();
    }
    if (tid == 0) {
        float mean = red[0] / (float)HW;
        float var  = (red2[0] - (float)HW * mean * mean) / (float)(HW - 1);
        bc[0] = mean;
        bc[1] = sqrtf(fmaxf(var, 0.f)) + 1e-6f;
    }
    __syncthreads();
    const float mean = bc[0], stdv = bc[1];
    for (int i = tid; i < HW; i += 256) {
        float v = raw_disp[(size_t)n * HW + i] * scale;
        disp_o[(size_t)n * HW + i]  = v;
        dnorm_o[(size_t)n * HW + i] = (v - mean) / stdv;
    }
}

// ---------------------------------------------------------------------------
// WMMA fragment helpers (16-bit A layout per CDNA5 ISA 7.12.2)
// ---------------------------------------------------------------------------
__device__ __forceinline__ v16h load_a_frag(const half_t* row, int koff, int kb) {
    v16h a;
#pragma unroll
    for (int jj = 0; jj < 8; ++jj) {
        int kk = koff + ((jj < 4) ? (kb + 2 * jj) : (16 + kb + 2 * (jj - 4)));
        hvec2 p = *(const hvec2*)(row + kk);
        a[2 * jj]     = p[0];
        a[2 * jj + 1] = p[1];
    }
    return a;
}

__device__ __forceinline__ v8f wmma_f16(v16h a, v16h b, v8f c) {
    return __builtin_amdgcn_wmma_f32_16x16x32_f16(false, a, false, b, (short)0, c, false, false);
}

// ---------------------------------------------------------------------------
// Hidden layer: K -> 256, ReLU, f16 result into LDS.
// Weight column panels (K x 16) are TDM-streamed into double-buffered LDS and
// shared by all 4 waves. Wave 0 drives the DMA; barriers fence buffer reuse.
// ---------------------------------------------------------------------------
template <int KDIM>
__device__ __forceinline__ void run_hidden(const half_t* __restrict__ Wg,   // [KDIM][256] f16
                                           const float* __restrict__ bias,  // [256]
                                           const half_t* aRow,              // this thread's A row
                                           int tileRow0, half_t* outL,      // LDS out, stride 256
                                           half_t* panels,                  // 2 x 4096 halves
                                           int lane, int lm, int hi, int kb) {
    if (threadIdx.x < 32) tdm_panel_load(Wg, lds_off(panels), KDIM, 64);
    for (int nt = 0; nt < 16; ++nt) {
        half_t* buf = panels + (nt & 1) * (256 * 16);
        if (threadIdx.x < 32) {
            __builtin_amdgcn_s_wait_tensorcnt(0);            // panel nt landed
            if (nt < 15)                                      // overlap next panel
                tdm_panel_load(Wg + (nt + 1) * 16,
                               lds_off(panels + ((nt + 1) & 1) * (256 * 16)), KDIM, 64);
        }
        __syncthreads();                                      // panel visible to all waves
        v8f acc = {};
#pragma unroll
        for (int ks = 0; ks < (KDIM >> 5); ++ks) {
            v16h a = load_a_frag(aRow, ks * 32, kb);
            v16h b;
#pragma unroll
            for (int s = 0; s < 16; ++s)
                b[s] = buf[(ks * 32 + ((s < 8) ? (kb + s) : (16 + kb + s - 8))) * 16 + lm];
            acc = wmma_f16(a, b, acc);
        }
        const float bv = bias[nt * 16 + lm];
#pragma unroll
        for (int r = 0; r < 8; ++r) {
            float v = fmaxf(acc[r] + bv, 0.0f);
            outL[(tileRow0 + (r + 8 * hi)) * 256 + nt * 16 + lm] = (half_t)v;
        }
        __syncthreads();                                      // buffer free for reuse
    }
}

// Output layer: 256 -> 16 (cols >= NOUT are zero-padded weights). Returns C frag.
__device__ __forceinline__ v8f run_out(const half_t* __restrict__ Wg,   // [256][16] f16
                                       const half_t* aRow, half_t* panels,
                                       int lane, int kb) {
    if (threadIdx.x < 32) {
        tdm_panel_load(Wg, lds_off(panels), 256, 4);          // contiguous 8KB panel
        __builtin_amdgcn_s_wait_tensorcnt(0);
    }
    __syncthreads();
    v8f acc = {};
#pragma unroll
    for (int ks = 0; ks < 8; ++ks) {
        v16h a = load_a_frag(aRow, ks * 32, kb);
        v16h b;
#pragma unroll
        for (int s = 0; s < 16; ++s)
            b[s] = panels[(ks * 32 + ((s < 8) ? (kb + s) : (16 + kb + s - 8))) * 16 + (lane & 15)];
        acc = wmma_f16(a, b, acc);
    }
    __syncthreads();                                          // panels free for next user
    return acc;
}

// ---------------------------------------------------------------------------
// K3: fused gather + posenc + dual WMMA MLP.
// 128 threads = 4 waves; 64 queries/block; each wave owns a 16-query tile.
// grid: (QQ/64, 4 offsets, N)
// ---------------------------------------------------------------------------
__global__ void __launch_bounds__(128)
fused_query(const float* __restrict__ coord, const float* __restrict__ cell,
            const float* __restrict__ feat,  const float* __restrict__ dnorm,
            const float* __restrict__ coefm, const float* __restrict__ freqm,
            const float* __restrict__ phase_w,
            const half_t* __restrict__ w1h, const float* __restrict__ im_b1,
            const half_t* __restrict__ w2h, const float* __restrict__ im_b2,
            const half_t* __restrict__ w3h, const float* __restrict__ im_b3,
            const half_t* __restrict__ d1h, const float* __restrict__ dp_b1,
            const half_t* __restrict__ d2h, const float* __restrict__ dp_b2,
            const half_t* __restrict__ d3h, const float* __restrict__ dp_b3,
            float* __restrict__ preds, float* __restrict__ pdisps,
            float* __restrict__ areas) {
    __shared__ half_t ximL[64 * 128];        // x_im activations (A layout rows)
    __shared__ half_t xdpL[64 * 96];         // x_dp activations (padded to 96)
    __shared__ half_t h1L[64 * 256];         // hidden ping
    __shared__ half_t h2L[64 * 256];         // hidden pong
    __shared__ half_t panelsL[2 * 256 * 16]; // TDM weight panel double buffer

    const int n = blockIdx.z;
    const int j = blockIdx.y;                       // offset index 0..3
    const float vx = (j < 2) ? -1.0f : 1.0f;
    const float vy = (j & 1) ? 1.0f : -1.0f;
    const int qbase = blockIdx.x * 64;

    // ---------------- Phase A: gather + build activations -------------------
    {
        const int qq = threadIdx.x >> 1;            // 0..63 (local query)
        const int hf = threadIdx.x & 1;             // channel half
        const int qglob = qbase + qq;
        const size_t cidx = ((size_t)n * QQ + qglob) * 2;
        const float c0 = coord[cidx + 0];
        const float c1 = coord[cidx + 1];
        const float cell0 = cell[cidx + 0];
        const float cell1 = cell[cidx + 1];
        const float rxy = 1.0f / 96.0f;

        float cy = clampf(c0 + vx * rxy + 1e-6f, -1.0f + 1e-6f, 1.0f - 1e-6f);
        float cx = clampf(c1 + vy * rxy + 1e-6f, -1.0f + 1e-6f, 1.0f - 1e-6f);
        int iy = (int)rintf(((cy + 1.0f) * 96.0f - 1.0f) * 0.5f);
        int ix = (int)rintf(((cx + 1.0f) * 96.0f - 1.0f) * 0.5f);
        iy = min(max(iy, 0), HH - 1);
        ix = min(max(ix, 0), WW - 1);
        const int gidx = iy * WW + ix;

        const float qcy = -1.0f + (2.0f * (float)iy + 1.0f) / 96.0f;
        const float qcx = -1.0f + (2.0f * (float)ix + 1.0f) / 96.0f;
        const float rel0 = (c0 - qcy) * 96.0f;
        const float rel1 = (c1 - qcx) * 96.0f;

        if (hf == 0)
            areas[((size_t)(n * 4 + j)) * QQ + qglob] = fabsf(rel0 * rel1) + 1e-9f;

        const float ph0 = cell0 * 96.0f, ph1 = cell1 * 96.0f;
        const size_t gbase = ((size_t)n * HD_) * HW + gidx;

        // x_im channels [64*hf, 64*hf+64)
        for (int t = 0; t < 64; ++t) {
            const int k = hf * 64 + t;
            const float r = (k < 64) ? rel0 : rel1;
            const float f = ldexpf(3.14159265358979323846f, k & 31);
            const float ang = r * f;
            const float sc = (((k >> 5) & 1) == 0) ? __sinf(ang) : __cosf(ang);
            const float qfreq = freqm[gbase + (size_t)k * HW];
            const float qcoef = coefm[gbase + (size_t)k * HW];
            const float ph = ph0 * phase_w[k] + ph1 * phase_w[HD_ + k];
            ximL[qq * 128 + k] = (half_t)(qcoef * (qfreq * sc + ph));
        }
        // x_dp channels [48*hf, 48*hf+48)
        for (int t = 0; t < 48; ++t) {
            const int c = hf * 48 + t;
            float v;
            if (c == 0)       v = dnorm[(size_t)n * HW + gidx];
            else if (c < 65)  v = feat[((size_t)(n * CIN + (c - 1))) * HW + gidx];
            else if (c == 65) v = rel0;
            else if (c == 66) v = rel1;
            else              v = 0.0f;
            xdpL[qq * 96 + c] = (half_t)v;
        }
    }
    __syncthreads();

    // ---------------- Phase B: WMMA MLPs ------------------------------------
    const int lane  = threadIdx.x & 31;
    const int wave  = threadIdx.x >> 5;             // 0..3: query tile
    const int lm    = lane & 15;                    // A: M row / C: N col
    const int hi    = lane >> 4;
    const int kb    = hi * 8;                       // A-layout K split
    const int tileRow0 = wave * 16;

    // ===== im MLP: 128 -> 256 -> 256 -> 3 =====
    run_hidden<128>(w1h, im_b1, &ximL[(tileRow0 + lm) * 128], tileRow0, h1L,
                    panelsL, lane, lm, hi, kb);
    run_hidden<256>(w2h, im_b2, &h1L[(tileRow0 + lm) * 256], tileRow0, h2L,
                    panelsL, lane, lm, hi, kb);
    {
        v8f acc = run_out(w3h, &h2L[(tileRow0 + lm) * 256], panelsL, lane, kb);
        if (lm < 3) {
            const float bias = im_b3[lm];
#pragma unroll
            for (int r = 0; r < 8; ++r) {
                int q = qbase + tileRow0 + (r + 8 * hi);
                preds[(((size_t)(n * 4 + j)) * QQ + q) * 3 + lm] = acc[r] + bias;
            }
        }
    }

    // ===== dp MLP: 96(pad of 67) -> 256 -> 256 -> 1 =====
    run_hidden<96>(d1h, dp_b1, &xdpL[(tileRow0 + lm) * 96], tileRow0, h1L,
                   panelsL, lane, lm, hi, kb);
    run_hidden<256>(d2h, dp_b2, &h1L[(tileRow0 + lm) * 256], tileRow0, h2L,
                    panelsL, lane, lm, hi, kb);
    {
        v8f acc = run_out(d3h, &h2L[(tileRow0 + lm) * 256], panelsL, lane, kb);
        if (lm == 0) {
            const float bias = dp_b3[0];
#pragma unroll
            for (int r = 0; r < 8; ++r) {
                int q = qbase + tileRow0 + (r + 8 * hi);
                pdisps[((size_t)(n * 4 + j)) * QQ + q] = acc[r] + bias;
            }
        }
    }
}

// ---------------------------------------------------------------------------
// K4: area-weighted combine + bilinear(border) of lr and disp -> output (N,Q,4)
// ---------------------------------------------------------------------------
__global__ void combine(const float* __restrict__ coord,
                        const float* __restrict__ lr, const float* __restrict__ disp,
                        const float* __restrict__ preds, const float* __restrict__ pdisps,
                        const float* __restrict__ areas, float* __restrict__ out) {
    const int t = blockIdx.x * blockDim.x + threadIdx.x;   // 0..N*QQ-1
    const int n = t / QQ;
    const int q = t % QQ;

    float a[4];
#pragma unroll
    for (int i = 0; i < 4; ++i) a[i] = areas[((size_t)(n * 4 + i)) * QQ + q];
    const float tot = a[0] + a[1] + a[2] + a[3];
    float wgt[4] = { a[3] / tot, a[2] / tot, a[1] / tot, a[0] / tot };  // order [3,2,1,0]

    float rgb0 = 0.f, rgb1 = 0.f, rgb2 = 0.f, pd = 0.f;
#pragma unroll
    for (int i = 0; i < 4; ++i) {
        const size_t pb = (((size_t)(n * 4 + i)) * QQ + q) * 3;
        rgb0 += preds[pb + 0] * wgt[i];
        rgb1 += preds[pb + 1] * wgt[i];
        rgb2 += preds[pb + 2] * wgt[i];
        pd   += pdisps[((size_t)(n * 4 + i)) * QQ + q] * wgt[i];
    }

    const float c0 = coord[((size_t)n * QQ + q) * 2 + 0];
    const float c1 = coord[((size_t)n * QQ + q) * 2 + 1];
    float fy = clampf(((c0 + 1.0f) * 96.0f - 1.0f) * 0.5f, 0.0f, 95.0f);
    float fx = clampf(((c1 + 1.0f) * 96.0f - 1.0f) * 0.5f, 0.0f, 95.0f);
    float y0 = floorf(fy), x0 = floorf(fx);
    float wy = fy - y0, wx = fx - x0;
    int y0i = min(max((int)y0, 0), HH - 1);
    int y1i = min(y0i + 1, HH - 1);
    int x0i = min(max((int)x0, 0), WW - 1);
    int x1i = min(x0i + 1, WW - 1);
    const float w00 = (1.f - wy) * (1.f - wx), w01 = (1.f - wy) * wx;
    const float w10 = wy * (1.f - wx), w11 = wy * wx;
    const int i00 = y0i * WW + x0i, i01 = y0i * WW + x1i;
    const int i10 = y1i * WW + x0i, i11 = y1i * WW + x1i;

    float rgb[3] = { rgb0, rgb1, rgb2 };
#pragma unroll
    for (int c = 0; c < 3; ++c) {
        const float* lp = lr + ((size_t)(n * 3 + c)) * HW;
        rgb[c] += lp[i00] * w00 + lp[i01] * w01 + lp[i10] * w10 + lp[i11] * w11;
    }
    const float* dp = disp + (size_t)n * HW;
    const float dv = dp[i00] * w00 + dp[i01] * w01 + dp[i10] * w10 + dp[i11] * w11;

    const size_t ob = ((size_t)n * QQ + q) * 4;
    out[ob + 0] = rgb[0];
    out[ob + 1] = rgb[1];
    out[ob + 2] = rgb[2];
    out[ob + 3] = pd + dv;
}

// ---------------------------------------------------------------------------
// Host launcher
// ---------------------------------------------------------------------------
extern "C" void kernel_launch(void* const* d_in, const int* in_sizes, int n_in,
                              void* d_out, int out_size, void* d_ws, size_t ws_size,
                              hipStream_t stream) {
    const float* lr       = (const float*)d_in[0];
    const float* feat     = (const float*)d_in[1];
    const float* raw_disp = (const float*)d_in[2];
    const float* coord    = (const float*)d_in[3];
    const float* cell     = (const float*)d_in[4];
    const float* coef_w   = (const float*)d_in[5];
    const float* coef_b   = (const float*)d_in[6];
    const float* freq_w   = (const float*)d_in[7];
    const float* freq_b   = (const float*)d_in[8];
    const float* phase_w  = (const float*)d_in[9];
    const float* im_w1    = (const float*)d_in[10];
    const float* im_b1    = (const float*)d_in[11];
    const float* im_w2    = (const float*)d_in[12];
    const float* im_b2    = (const float*)d_in[13];
    const float* im_w3    = (const float*)d_in[14];
    const float* im_b3    = (const float*)d_in[15];
    const float* dp_w1    = (const float*)d_in[16];
    const float* dp_b1    = (const float*)d_in[17];
    const float* dp_w2    = (const float*)d_in[18];
    const float* dp_b2    = (const float*)d_in[19];
    const float* dp_w3    = (const float*)d_in[20];
    const float* dp_b3    = (const float*)d_in[21];
    float* out = (float*)d_out;

    // Workspace carving (256B aligned)
    char* base = (char*)d_ws;
    size_t off = 0;
    auto carve = [&](size_t bytes) -> char* {
        char* p = base + off;
        off = (off + bytes + 255) & ~(size_t)255;
        return p;
    };
    float*  coefws  = (float*) carve((size_t)NB * HD_ * HW * 4);
    float*  freqws  = (float*) carve((size_t)NB * HD_ * HW * 4);
    float*  dispws  = (float*) carve((size_t)NB * HW * 4);
    float*  dnormws = (float*) carve((size_t)NB * HW * 4);
    half_t* w1h     = (half_t*)carve(128 * 256 * 2);
    half_t* w2h     = (half_t*)carve(256 * 256 * 2);
    half_t* w3h     = (half_t*)carve(256 * 16 * 2);
    half_t* d1h     = (half_t*)carve(96 * 256 * 2);
    half_t* d2h     = (half_t*)carve(256 * 256 * 2);
    half_t* d3h     = (half_t*)carve(256 * 16 * 2);
    float*  predsws = (float*) carve((size_t)NB * 4 * QQ * 3 * 4);
    float*  pdispws = (float*) carve((size_t)NB * 4 * QQ * 4);
    float*  areaws  = (float*) carve((size_t)NB * 4 * QQ * 4);
    (void)ws_size; (void)in_sizes; (void)n_in; (void)out_size;

    prep_weights<<<64, 256, 0, stream>>>(im_w1, im_w2, im_w3, dp_w1, dp_w2, dp_w3,
                                         w1h, w2h, w3h, d1h, d2h, d3h);

    conv3x3<<<dim3(HW / 256, HD_, NB), 256, 0, stream>>>(feat, coef_w, coef_b,
                                                         freq_w, freq_b, coefws, freqws);

    disp_stats<<<NB, 256, 0, stream>>>(raw_disp, cell, dispws, dnormws);

    fused_query<<<dim3(QQ / 64, 4, NB), 128, 0, stream>>>(
        coord, cell, feat, dnormws, coefws, freqws, phase_w,
        w1h, im_b1, w2h, im_b2, w3h, im_b3,
        d1h, dp_b1, d2h, dp_b2, d3h, dp_b3,
        predsws, pdispws, areaws);

    combine<<<(NB * QQ) / 256, 256, 0, stream>>>(coord, lr, dispws,
                                                 predsws, pdispws, areaws, out);
}
